// SelfAttention_82583631167780
// MI455X (gfx1250) — compile-verified, matrix operand order checked by reading, not run on hardware
//
#include <hip/hip_runtime.h>
#include <hip/hip_bf16.h>

// Problem sizes (fixed by the reference)
#define BB 4
#define SS 2048
#define DD 1024
#define MM (BB * SS)   // 8192 total query rows

typedef __attribute__((ext_vector_type(16))) __bf16 v16bf;
typedef __attribute__((ext_vector_type(8)))  __bf16 v8bf;
typedef __attribute__((ext_vector_type(8)))  float  v8f;

// ---------------------------------------------------------------------------
// WMMA helpers (CDNA5 wave32, V_WMMA_F32_16X16X32_BF16)
// A-fragment layout (16-bit A 16x32): lane L -> row m = L&15, half = L>>4;
// elements 0..7  = K[half*8 .. half*8+7], elements 8..15 = K[16+half*8 ..].
// B-fragment is symmetric with n = L&15 (we always feed "NT" operands whose
// K is contiguous in memory), so one loader serves both operands.
// ---------------------------------------------------------------------------
__device__ __forceinline__ v16bf frag_nt(const __bf16* base, int ld) {
  const int lane = threadIdx.x & 31;
  const __bf16* p = base + (size_t)(lane & 15) * ld + ((lane >> 4) << 3);
  v16bf f;
  ((v8bf*)&f)[0] = *(const v8bf*)(p);        // 16B: K chunk at half*8
  ((v8bf*)&f)[1] = *(const v8bf*)(p + 16);   // 16B: K chunk at 16+half*8
  return f;
}

__device__ __forceinline__ v8f wmma_bf16(v16bf a, v16bf b, v8f c) {
  return __builtin_amdgcn_wmma_f32_16x16x32_bf16(
      /*neg_a=*/false, a, /*neg_b=*/false, b,
      /*c_mod=*/(short)0, c, /*reuse_a=*/false, /*reuse_b=*/false);
}

// ---------------------------------------------------------------------------
// fp32 -> bf16 elementwise convert (for x)
// ---------------------------------------------------------------------------
__global__ void cvt_f32_bf16(const float* __restrict__ src,
                             __bf16* __restrict__ dst, int n) {
  int i = blockIdx.x * blockDim.x + threadIdx.x;
  const int stride = gridDim.x * blockDim.x;
  for (; i < n; i += stride) dst[i] = (__bf16)src[i];
}

// ---------------------------------------------------------------------------
// Weight transpose + convert: W[k, n] (f32, DxD) -> WT[n, k] (bf16)
// ---------------------------------------------------------------------------
__global__ void wtrans_f32_bf16(const float* __restrict__ W,
                                __bf16* __restrict__ WT) {
  const int i = blockIdx.x * blockDim.x + threadIdx.x;  // over D*D
  const int k = i / DD;
  const int n = i - k * DD;
  WT[(size_t)n * DD + k] = (__bf16)W[i];
}

// ---------------------------------------------------------------------------
// Projection GEMM: OUT[m,n] = sum_k X[m,k] * WT[n,k]
//   M = B*S = 8192, K = N = D = 1024.
// Block = 256 threads (8 waves), block tile 128x128; wave w owns rows
// [16w,16w+16) x 128 cols (8 independent accumulator chains, A reused 8x).
// OUT_MODE 0: row-major bf16 [M,N] (Q, K).
// OUT_MODE 1: per-batch transposed bf16 VT[b][n][s] so P*V becomes NT;
//             the 8 accumulator rows are contiguous -> one 16B store.
// ---------------------------------------------------------------------------
template <int OUT_MODE>
__global__ void __launch_bounds__(256) proj_gemm(const __bf16* __restrict__ X,
                                                 const __bf16* __restrict__ WT,
                                                 __bf16* __restrict__ OUT) {
  const int wave = threadIdx.x >> 5;
  const int lane = threadIdx.x & 31;
  const int row0 = blockIdx.x * 128 + wave * 16;
  const int col0 = blockIdx.y * 128;

  v8f acc[8] = {};
  for (int k0 = 0; k0 < DD; k0 += 32) {
    const v16bf a = frag_nt(X + (size_t)row0 * DD + k0, DD);
#pragma unroll
    for (int j = 0; j < 8; ++j) {
      const v16bf b = frag_nt(WT + (size_t)(col0 + j * 16) * DD + k0, DD);
      acc[j] = wmma_bf16(a, b, acc[j]);
    }
  }

  const int half = lane >> 4, nl = lane & 15;
  if (OUT_MODE == 0) {
#pragma unroll
    for (int j = 0; j < 8; ++j)
#pragma unroll
      for (int r = 0; r < 8; ++r)
        OUT[(size_t)(row0 + 8 * half + r) * DD + (col0 + j * 16 + nl)] =
            (__bf16)acc[j][r];
  } else {
    const int m0 = row0 + 8 * half;          // 8 consecutive rows start here
    const int b  = m0 / SS;
    const int s0 = m0 - b * SS;
#pragma unroll
    for (int j = 0; j < 8; ++j) {
      alignas(16) __bf16 tmp[8];
#pragma unroll
      for (int r = 0; r < 8; ++r) tmp[r] = (__bf16)acc[j][r];
      *(v8bf*)(OUT + ((size_t)b * DD + (col0 + j * 16 + nl)) * SS + s0) =
          *(const v8bf*)tmp;
    }
  }
}

// ---------------------------------------------------------------------------
// Flash-style attention. One wave owns a 16-row query tile.
// Block = 2 waves (64 threads), dynamic LDS = 2 * 16 * S bf16 = 128 KB.
// Phase 1: scores S[16,S] = Q_tile . K^T via WMMA (two 16-key accumulator
//          chains per iteration sharing each Q fragment; the compiler hoists
//          the whole 16x1024 Q tile into high VGPRs), staged raw in LDS.
// Phase 2: per-row max & expsum via LDS rescan + __shfl_xor (16-lane halves),
//          exp written back in place; 1/l folded into the output store.
// Phase 3: O_tile = P . V via WMMA against VT (NT); two 16-col chains share
//          each LDS P fragment (halves ds traffic, doubles ILP), f32 stores.
// ---------------------------------------------------------------------------
#define ATTN_WAVES 2

__global__ void __launch_bounds__(ATTN_WAVES * 32) attn_kernel(
    const __bf16* __restrict__ Q, const __bf16* __restrict__ K,
    const __bf16* __restrict__ VT, float* __restrict__ out) {
  extern __shared__ __bf16 sP[];  // [ATTN_WAVES][16][SS]
  const int wave = threadIdx.x >> 5;
  const int lane = threadIdx.x & 31;
  const int half = lane >> 4, nl = lane & 15;

  const int tile = blockIdx.x * ATTN_WAVES + wave;  // 0 .. MM/16-1
  const int b    = tile >> 7;                       // / (SS/16)
  const int q0   = (tile & 127) << 4;

  const __bf16* Qb  = Q  + ((size_t)b * SS + q0) * DD;
  const __bf16* Kb  = K  + (size_t)b * SS * DD;
  const __bf16* VTb = VT + (size_t)b * DD * SS;
  __bf16* myP = sP + (size_t)wave * 16 * SS;

  const float scale = 0.03125f;  // 1/sqrt(1024)

  // ---- Phase 1: raw scaled scores -> LDS (two WMMA chains / iteration) ----
  for (int k0 = 0; k0 < SS; k0 += 32) {
    // Prefetch the upcoming key rows (lane-spread -> distinct cachelines).
    if (k0 + 32 < SS)
      __builtin_prefetch(Kb + (size_t)(k0 + 32 + lane) * DD, 0, 1);
    v8f acc0 = {0.f, 0.f, 0.f, 0.f, 0.f, 0.f, 0.f, 0.f};
    v8f acc1 = {0.f, 0.f, 0.f, 0.f, 0.f, 0.f, 0.f, 0.f};
    for (int d0 = 0; d0 < DD; d0 += 32) {
      const v16bf a   = frag_nt(Qb + d0, DD);                          // q rows
      const v16bf kb0 = frag_nt(Kb + (size_t)k0 * DD + d0, DD);        // keys
      const v16bf kb1 = frag_nt(Kb + (size_t)(k0 + 16) * DD + d0, DD);
      acc0 = wmma_bf16(a, kb0, acc0);
      acc1 = wmma_bf16(a, kb1, acc1);
    }
#pragma unroll
    for (int r = 0; r < 8; ++r) {
      __bf16* rowp = myP + (size_t)(r + 8 * half) * SS + k0 + nl;
      rowp[0]  = (__bf16)(acc0[r] * scale);
      rowp[16] = (__bf16)(acc1[r] * scale);
    }
  }

  // ---- Phase 2: softmax stats + exp in place ----
  // Row m = r + 8*half is processed by the 16 lanes of its half-wave.
  float linv[8];
#pragma unroll
  for (int r = 0; r < 8; ++r) {
    __bf16* rowp = myP + (size_t)(r + 8 * half) * SS;
    float mx = -1e30f;
    for (int j = nl; j < SS; j += 16) mx = fmaxf(mx, (float)rowp[j]);
    mx = fmaxf(mx, __shfl_xor(mx, 1, 32));
    mx = fmaxf(mx, __shfl_xor(mx, 2, 32));
    mx = fmaxf(mx, __shfl_xor(mx, 4, 32));
    mx = fmaxf(mx, __shfl_xor(mx, 8, 32));
    float sum = 0.f;
    for (int j = nl; j < SS; j += 16) {
      const float e = __expf((float)rowp[j] - mx);
      rowp[j] = (__bf16)e;
      sum += e;
    }
    sum += __shfl_xor(sum, 1, 32);
    sum += __shfl_xor(sum, 2, 32);
    sum += __shfl_xor(sum, 4, 32);
    sum += __shfl_xor(sum, 8, 32);
    linv[r] = 1.0f / sum;
  }
  __syncthreads();

  // ---- Phase 3: O = P . V (NT against VT); two chains share each P frag ----
  for (int d0 = 0; d0 < DD; d0 += 32) {
    v8f acc0 = {0.f, 0.f, 0.f, 0.f, 0.f, 0.f, 0.f, 0.f};
    v8f acc1 = {0.f, 0.f, 0.f, 0.f, 0.f, 0.f, 0.f, 0.f};
    for (int k0 = 0; k0 < SS; k0 += 32) {
      const v16bf a   = frag_nt(myP + k0, SS);                         // P (LDS)
      const v16bf vb0 = frag_nt(VTb + (size_t)d0 * SS + k0, SS);       // V^T
      const v16bf vb1 = frag_nt(VTb + (size_t)(d0 + 16) * SS + k0, SS);
      acc0 = wmma_bf16(a, vb0, acc0);
      acc1 = wmma_bf16(a, vb1, acc1);
    }
    float* op = out + ((size_t)b * SS + q0 + 8 * half) * DD + d0 + nl;
#pragma unroll
    for (int r = 0; r < 8; ++r) {
      op[(size_t)r * DD]      = acc0[r] * linv[r];
      op[(size_t)r * DD + 16] = acc1[r] * linv[r];
    }
  }
}

// ---------------------------------------------------------------------------
// Host-side orchestration.
// Workspace layout (bf16): xb[M*D] | WqT | WkT | WvT (D*D each) |
//                          Qb[M*D] | Kb[M*D] | VT[B*D*S]   -> ~70 MB total.
// ---------------------------------------------------------------------------
extern "C" void kernel_launch(void* const* d_in, const int* in_sizes, int n_in,
                              void* d_out, int out_size, void* d_ws,
                              size_t ws_size, hipStream_t stream) {
  (void)in_sizes; (void)n_in; (void)out_size; (void)ws_size;
  const float* x  = (const float*)d_in[0];
  const float* Wq = (const float*)d_in[1];
  const float* Wk = (const float*)d_in[2];
  const float* Wv = (const float*)d_in[3];
  float* out = (float*)d_out;

  const size_t nX = (size_t)MM * DD;   // 8 Mi elements
  const size_t nW = (size_t)DD * DD;   // 1 Mi elements
  char* ws = (char*)d_ws;
  __bf16* xb  = (__bf16*)(ws);
  __bf16* WqT = xb  + nX;
  __bf16* WkT = WqT + nW;
  __bf16* WvT = WkT + nW;
  __bf16* Qb  = WvT + nW;
  __bf16* Kb  = Qb  + nX;
  __bf16* VT  = Kb  + nX;

  cvt_f32_bf16<<<4096, 256, 0, stream>>>(x, xb, (int)nX);
  wtrans_f32_bf16<<<(int)(nW / 256), 256, 0, stream>>>(Wq, WqT);
  wtrans_f32_bf16<<<(int)(nW / 256), 256, 0, stream>>>(Wk, WkT);
  wtrans_f32_bf16<<<(int)(nW / 256), 256, 0, stream>>>(Wv, WvT);

  dim3 gproj(MM / 128, DD / 128);  // (64, 8)
  proj_gemm<0><<<gproj, 256, 0, stream>>>(xb, WqT, Qb);
  proj_gemm<0><<<gproj, 256, 0, stream>>>(xb, WkT, Kb);
  proj_gemm<1><<<gproj, 256, 0, stream>>>(xb, WvT, VT);

  const size_t ldsBytes = (size_t)ATTN_WAVES * 16 * SS * sizeof(__bf16);  // 128 KB
  attn_kernel<<<MM / 16 / ATTN_WAVES, ATTN_WAVES * 32, ldsBytes, stream>>>(
      Qb, Kb, VT, out);
}
